// Block_4200478015844
// MI455X (gfx1250) — matrix operand order repaired and checked by reading, not compiled, not run on hardware
//
#include <hip/hip_runtime.h>
#include <math.h>

typedef __bf16 bf16;
typedef __attribute__((ext_vector_type(16))) __bf16 v16bf;
typedef __attribute__((ext_vector_type(8)))  __bf16 v8bf;
typedef __attribute__((ext_vector_type(8)))  float  v8f;

#define B_    2
#define T_    2048
#define C_    2048
#define H_    16
#define G_    4
#define HS_   128
#define DFF_  8192
#define MROWS (B_*T_)            // 4096
#define NQKV  ((H_+2*G_)*HS_)    // 3072

// ---------- CDNA5 async global->LDS (ASYNCcnt-tracked) ----------
static __device__ __forceinline__ void async_load_b128(const bf16* lds_dst, const bf16* gsrc) {
  // low 32 bits of a flat shared pointer are the LDS byte offset
  unsigned lds_addr = (unsigned)(size_t)lds_dst;
  unsigned long long gaddr = (unsigned long long)(size_t)gsrc;
  asm volatile("global_load_async_to_lds_b128 %0, %1, off"
               :: "v"(lds_addr), "v"(gaddr) : "memory");
}
static __device__ __forceinline__ void wait_async0() {
  asm volatile("s_wait_asynccnt 0x0" ::: "memory");
}

// ---------- 16-lane row reductions (wave32 half-wave groups) ----------
static __device__ __forceinline__ float rowmax16(float x) {
#pragma unroll
  for (int o = 1; o < 16; o <<= 1) x = fmaxf(x, __shfl_xor(x, o, 32));
  return x;
}
static __device__ __forceinline__ float rowsum16(float x) {
#pragma unroll
  for (int o = 1; o < 16; o <<= 1) x += __shfl_xor(x, o, 32);
  return x;
}

// ---------- f32 -> bf16 conversion ----------
__global__ __launch_bounds__(256) void f32_to_bf16_kernel(const float* __restrict__ in,
                                                          bf16* __restrict__ out, int n) {
  int i = blockIdx.x * 256 + threadIdx.x;
  if (i < n) out[i] = (bf16)in[i];
}

// ---------- LayerNorm (one block per row of 2048) ----------
__global__ __launch_bounds__(256) void layernorm_kernel(const float* __restrict__ x,
                                                        const float* __restrict__ w,
                                                        const float* __restrict__ b,
                                                        bf16* __restrict__ out) {
  __shared__ float red[256];
  const int tid = threadIdx.x;
  const size_t row = blockIdx.x;
  const float* xr = x + row * C_;
  float4 v0 = ((const float4*)xr)[tid * 2];
  float4 v1 = ((const float4*)xr)[tid * 2 + 1];
  float vals[8] = {v0.x, v0.y, v0.z, v0.w, v1.x, v1.y, v1.z, v1.w};
  float s = 0.f;
#pragma unroll
  for (int e = 0; e < 8; ++e) s += vals[e];
  red[tid] = s; __syncthreads();
  for (int st = 128; st > 0; st >>= 1) {
    if (tid < st) red[tid] += red[tid + st];
    __syncthreads();
  }
  float mu = red[0] * (1.0f / C_);
  __syncthreads();
  float q = 0.f;
#pragma unroll
  for (int e = 0; e < 8; ++e) { float d = vals[e] - mu; q += d * d; }
  red[tid] = q; __syncthreads();
  for (int st = 128; st > 0; st >>= 1) {
    if (tid < st) red[tid] += red[tid + st];
    __syncthreads();
  }
  float var = red[0] * (1.0f / C_);
  float rs = rsqrtf(var + 1e-5f);
  int c0 = tid * 8;
#pragma unroll
  for (int e = 0; e < 8; ++e)
    out[row * C_ + c0 + e] = (bf16)((vals[e] - mu) * rs * w[c0 + e] + b[c0 + e]);
}

// ---------- Tiled WMMA GEMM:  out[M,N] = A[M,K](bf16) * W[N,K]^T(bf16) ----------
// 128x128 block tile, 4 waves, 64x64 per wave (16 WMMAs / 16 ds_load_b128 per K-step).
// Tiles filled with global_load_async_to_lds_b128, double-buffered on ASYNCcnt.
// MODE 0: outF = acc              (QKV)
// MODE 1: outF = acc + resid      (proj + residual)
// MODE 2: outB = gelu(acc+bias)   (fc1)
// MODE 3: outF = acc+bias+resid   (fc2 + residual -> final output)
template <int MODE>
__global__ __launch_bounds__(128) void gemm_bf16_kernel(const bf16* __restrict__ A,
                                                        const bf16* __restrict__ W,
                                                        int M, int N, int K,
                                                        float* __restrict__ outF,
                                                        bf16* __restrict__ outB,
                                                        const float* __restrict__ bias,
                                                        const float* __restrict__ resid) {
  __shared__ bf16 As[2][128 * 40];   // 128 rows x 32 K, stride 40 (bank stagger, rows 16B-aligned)
  __shared__ bf16 Ws[2][128 * 40];
  const int tid  = threadIdx.x;
  const int wave = tid >> 5, lane = tid & 31;
  const int lo = lane & 15, hi = lane >> 4;
  const int wm = (wave >> 1) * 64;   // wave's M offset in tile
  const int wn = (wave & 1) * 64;    // wave's N offset in tile
  const int m0 = blockIdx.y * 128, n0 = blockIdx.x * 128;

  v8f acc[4][4] = {};

  auto loadTiles = [&](int kt, int buf) {
    const int k0 = kt * 32;
#pragma unroll
    for (int i = 0; i < 4; ++i) {
      int c = tid + i * 128;               // 512 chunks of 8 bf16 per array
      int r = c >> 2, k8 = (c & 3) * 8;
      async_load_b128(&As[buf][r * 40 + k8], A + (size_t)(m0 + r) * K + k0 + k8);
      async_load_b128(&Ws[buf][r * 40 + k8], W + (size_t)(n0 + r) * K + k0 + k8);
    }
  };

  loadTiles(0, 0);
  wait_async0();
  __syncthreads();
  const int nk = K >> 5;
  for (int kt = 0; kt < nk; ++kt) {
    const int cur = kt & 1;
    if (kt + 1 < nk) loadTiles(kt + 1, cur ^ 1);

    union { v16bf v; v8bf h2[2]; } afr[4];
#pragma unroll
    for (int mt = 0; mt < 4; ++mt) {
      const bf16* p = &As[cur][(wm + mt * 16 + lo) * 40 + hi * 8];
      afr[mt].h2[0] = *(const v8bf*)p;          // K = c0..c0+7
      afr[mt].h2[1] = *(const v8bf*)(p + 16);   // K = c0+16..c0+23
    }
#pragma unroll
    for (int nt = 0; nt < 4; ++nt) {
      union { v16bf v; v8bf h2[2]; } bfr;
      const bf16* p = &Ws[cur][(wn + nt * 16 + lo) * 40 + hi * 8];
      bfr.h2[0] = *(const v8bf*)p;
      bfr.h2[1] = *(const v8bf*)(p + 16);
#pragma unroll
      for (int mt = 0; mt < 4; ++mt)
        acc[mt][nt] = __builtin_amdgcn_wmma_f32_16x16x32_bf16(
            false, afr[mt].v, false, bfr.v, (short)0, acc[mt][nt], false, false);
    }
    wait_async0();     // next tile fully deposited in LDS
    __syncthreads();
  }

#pragma unroll
  for (int mt = 0; mt < 4; ++mt)
#pragma unroll
    for (int nt = 0; nt < 4; ++nt)
#pragma unroll
      for (int v = 0; v < 8; ++v) {
        int m = m0 + wm + mt * 16 + v + hi * 8;   // C-fragment: M = vgpr + 8*(lane/16)
        int n = n0 + wn + nt * 16 + lo;           //             N = lane%16
        float r = acc[mt][nt][v];
        size_t idx = (size_t)m * N + n;
        if (MODE == 0) outF[idx] = r;
        else if (MODE == 1) outF[idx] = r + resid[idx];
        else if (MODE == 2) {
          float t = r + bias[n];
          outB[idx] = (bf16)(0.5f * t * (1.0f + erff(t * 0.70710678118654752f)));
        } else outF[idx] = r + bias[n] + resid[idx];
      }
}

// ---------- RoPE + scale + GQA split: qkv f32 -> Q/K/V bf16 ----------
__global__ __launch_bounds__(256) void rope_split_kernel(const float* __restrict__ qkv,
                                                         const float* __restrict__ cosb,
                                                         const float* __restrict__ sinb,
                                                         bf16* __restrict__ Qd,
                                                         bf16* __restrict__ Kd,
                                                         bf16* __restrict__ Vd) {
  int idx  = blockIdx.x * 256 + threadIdx.x;      // exactly B*T*3072 threads
  int col  = idx % NQKV;
  int row  = idx / NQKV;
  int b = row / T_, t = row % T_;
  int g = col / (6 * HS_);
  int r = col % (6 * HS_);
  int slot = r / HS_, hs = r % HS_;
  float x = qkv[(size_t)row * NQKV + col];
  if (slot == 5) {                                // V: passthrough
    Vd[((size_t)(b * G_ + g) * T_ + t) * HS_ + hs] = (bf16)x;
    return;
  }
  float c = cosb[t * HS_ + hs], s = sinb[t * HS_ + hs];
  float partner = (hs < HS_ / 2) ? -qkv[(size_t)row * NQKV + col + HS_ / 2]
                                 :  qkv[(size_t)row * NQKV + col - HS_ / 2];
  float o = x * c + partner * s;
  if (slot == 4) {
    Kd[((size_t)(b * G_ + g) * T_ + t) * HS_ + hs] = (bf16)o;
  } else {
    int h = g * (H_ / G_) + slot;
    Qd[((size_t)(b * H_ + h) * T_ + t) * HS_ + hs] = (bf16)(o * 0.08838834764831845f); // 1/sqrt(128)
  }
}

// ---------- Flash attention: 64-query tile per block, 4 waves ----------
__global__ __launch_bounds__(128) void attention_kernel(const bf16* __restrict__ Q,
                                                        const bf16* __restrict__ Kg,
                                                        const bf16* __restrict__ Vg,
                                                        bf16* __restrict__ Y) {
  __shared__ bf16 Qs[64 * 136];       // 64 q rows x 128 hs (pad 8)
  __shared__ bf16 Ks[64 * 136];       // 64 k rows x 128 hs
  __shared__ bf16 Vt[128 * 72];       // transposed: 128 hs x 64 keys (pad 8)
  __shared__ bf16 Ps[4 * 16 * 72];    // per-wave P strip: 16 q x 64 keys

  const int tid = threadIdx.x;
  const int wave = tid >> 5, lane = tid & 31;
  const int lo = lane & 15, hi = lane >> 4;
  const int qtile = blockIdx.x;
  const int bh = blockIdx.y;
  const int b = bh / H_, h = bh % H_;
  const int bg = b * G_ + h / (H_ / G_);
  const int q0 = qtile * 64;
  const int wq = wave * 16;

  // load Q tile async (stays resident)
  const bf16* Qbase = Q + ((size_t)bh * T_ + q0) * HS_;
#pragma unroll
  for (int i = 0; i < 8; ++i) {
    int c = tid + i * 128;
    int r = c >> 4, k8 = (c & 15) * 8;
    async_load_b128(&Qs[r * 136 + k8], Qbase + (size_t)r * HS_ + k8);
  }

  float mrow[8], lrow[8];
#pragma unroll
  for (int v = 0; v < 8; ++v) { mrow[v] = -1e30f; lrow[v] = 0.0f; }
  v8f Of[8] = {};
  bf16* myP = &Ps[wave * 16 * 72];

  for (int j = 0; j <= qtile; ++j) {     // causal: only kv tiles <= q tile
    const int kv0 = j * 64;
    __syncthreads();
    const bf16* Kbase = Kg + ((size_t)bg * T_ + kv0) * HS_;
    const bf16* Vbase = Vg + ((size_t)bg * T_ + kv0) * HS_;
#pragma unroll
    for (int i = 0; i < 8; ++i) {
      int c = tid + i * 128;
      int r = c >> 4, k8 = (c & 15) * 8;
      async_load_b128(&Ks[r * 136 + k8], Kbase + (size_t)r * HS_ + k8);
      v8bf vv = *(const v8bf*)(Vbase + (size_t)r * HS_ + k8);
#pragma unroll
      for (int e = 0; e < 8; ++e) Vt[(k8 + e) * 72 + r] = vv[e];  // transpose into LDS
    }
    wait_async0();           // Q (first iter) + K tile deposited
    __syncthreads();

    // S = Q * K^T   (wave's 16 q rows x 64 keys), K-dim = 128 -> 4 wmma steps
    v8f Sf[4] = {};
#pragma unroll
    for (int ks = 0; ks < 4; ++ks) {
      union { v16bf v; v8bf h2[2]; } af;
      const bf16* ap = &Qs[(wq + lo) * 136 + ks * 32 + hi * 8];
      af.h2[0] = *(const v8bf*)ap;
      af.h2[1] = *(const v8bf*)(ap + 16);
#pragma unroll
      for (int nt = 0; nt < 4; ++nt) {
        union { v16bf v; v8bf h2[2]; } bf_;
        const bf16* bp = &Ks[(nt * 16 + lo) * 136 + ks * 32 + hi * 8];
        bf_.h2[0] = *(const v8bf*)bp;
        bf_.h2[1] = *(const v8bf*)(bp + 16);
        Sf[nt] = __builtin_amdgcn_wmma_f32_16x16x32_bf16(
            false, af.v, false, bf_.v, (short)0, Sf[nt], false, false);
      }
    }

    if (kv0 == q0) {  // diagonal tile: mask key > query
#pragma unroll
      for (int nt = 0; nt < 4; ++nt)
#pragma unroll
        for (int v = 0; v < 8; ++v)
          if (nt * 16 + lo > wq + v + hi * 8) Sf[nt][v] = -1e30f;
    }

    // online softmax (row = v + 8*hi, replicated across the 16-lane group)
    float sc[8];
#pragma unroll
    for (int v = 0; v < 8; ++v) {
      float mx = fmaxf(fmaxf(Sf[0][v], Sf[1][v]), fmaxf(Sf[2][v], Sf[3][v]));
      mx = rowmax16(mx);
      float mnew = fmaxf(mrow[v], mx);
      sc[v] = __expf(mrow[v] - mnew);
      float rs = 0.0f;
#pragma unroll
      for (int nt = 0; nt < 4; ++nt) {
        float p = __expf(Sf[nt][v] - mnew);
        Sf[nt][v] = p;
        rs += p;
      }
      lrow[v] = lrow[v] * sc[v] + rowsum16(rs);
      mrow[v] = mnew;
    }
#pragma unroll
    for (int f = 0; f < 8; ++f)
#pragma unroll
      for (int v = 0; v < 8; ++v) Of[f][v] *= sc[v];

    // C-fragment -> A-fragment layout change through per-wave LDS strip
#pragma unroll
    for (int nt = 0; nt < 4; ++nt)
#pragma unroll
      for (int v = 0; v < 8; ++v)
        myP[(v + hi * 8) * 72 + nt * 16 + lo] = (bf16)Sf[nt][v];

    // O += P * V   (K-dim = 64 keys -> 2 wmma steps, 8 hs tiles)
#pragma unroll
    for (int ks = 0; ks < 2; ++ks) {
      union { v16bf v; v8bf h2[2]; } af;
      const bf16* ap = &myP[lo * 72 + ks * 32 + hi * 8];
      af.h2[0] = *(const v8bf*)ap;
      af.h2[1] = *(const v8bf*)(ap + 16);
#pragma unroll
      for (int f = 0; f < 8; ++f) {
        union { v16bf v; v8bf h2[2]; } bf_;
        const bf16* bp = &Vt[(f * 16 + lo) * 72 + ks * 32 + hi * 8];
        bf_.h2[0] = *(const v8bf*)bp;
        bf_.h2[1] = *(const v8bf*)(bp + 16);
        Of[f] = __builtin_amdgcn_wmma_f32_16x16x32_bf16(
            false, af.v, false, bf_.v, (short)0, Of[f], false, false);
      }
    }
  }

  // normalize and store y[B*T, H*HS] (row-major -> proj GEMM A matrix)
#pragma unroll
  for (int v = 0; v < 8; ++v) {
    float linv = 1.0f / lrow[v];
    int qr = q0 + wq + v + hi * 8;
#pragma unroll
    for (int f = 0; f < 8; ++f)
      Y[((size_t)(b * T_) + qr) * C_ + h * HS_ + f * 16 + lo] = (bf16)(Of[f][v] * linv);
  }
}

// ---------- host orchestration ----------
extern "C" void kernel_launch(void* const* d_in, const int* in_sizes, int n_in,
                              void* d_out, int out_size, void* d_ws, size_t ws_size,
                              hipStream_t stream) {
  (void)in_sizes; (void)n_in; (void)out_size; (void)ws_size;
  const float* x      = (const float*)d_in[0];
  const float* cosb   = (const float*)d_in[1];
  const float* sinb   = (const float*)d_in[2];
  /* d_in[3] = mask: causal handled analytically */
  const float* ln1_w  = (const float*)d_in[4];
  const float* ln1_b  = (const float*)d_in[5];
  const float* qkv_w  = (const float*)d_in[6];
  const float* proj_w = (const float*)d_in[7];
  const float* ln2_w  = (const float*)d_in[8];
  const float* ln2_b  = (const float*)d_in[9];
  const float* fc1_w  = (const float*)d_in[10];
  const float* fc1_b  = (const float*)d_in[11];
  const float* fc2_w  = (const float*)d_in[12];
  const float* fc2_b  = (const float*)d_in[13];
  float* out = (float*)d_out;

  char* ws = (char*)d_ws;
  size_t off = 0;
  auto alloc = [&](size_t bytes) {
    char* p = ws + off;
    off += (bytes + 255) & ~(size_t)255;
    return p;
  };
  bf16*  wqkv  = (bf16*)alloc((size_t)NQKV * C_ * 2);
  bf16*  wproj = (bf16*)alloc((size_t)C_ * C_ * 2);
  bf16*  wfc1  = (bf16*)alloc((size_t)DFF_ * C_ * 2);
  bf16*  wfc2  = (bf16*)alloc((size_t)C_ * DFF_ * 2);
  bf16*  xn    = (bf16*)alloc((size_t)MROWS * C_ * 2);
  float* qkvF  = (float*)alloc((size_t)MROWS * NQKV * 4);       // reused for mlp
  bf16*  Qb    = (bf16*)alloc((size_t)B_ * H_ * T_ * HS_ * 2);  // contiguous after qkvF
  bf16*  Kb    = (bf16*)alloc((size_t)B_ * G_ * T_ * HS_ * 2);
  bf16*  Vb    = (bf16*)alloc((size_t)B_ * G_ * T_ * HS_ * 2);
  bf16*  y     = (bf16*)alloc((size_t)MROWS * C_ * 2);
  float* x1    = (float*)alloc((size_t)MROWS * C_ * 4);
  bf16*  mlp   = (bf16*)qkvF;   // aliases qkvF+Qb (67.1 MB), dead by then
  bf16*  xn2   = xn;            // xn dead after QKV GEMM

  auto conv = [&](const float* src, bf16* dst, size_t n) {
    f32_to_bf16_kernel<<<(unsigned)((n + 255) / 256), 256, 0, stream>>>(src, dst, (int)n);
  };
  conv(qkv_w,  wqkv,  (size_t)NQKV * C_);
  conv(proj_w, wproj, (size_t)C_ * C_);
  conv(fc1_w,  wfc1,  (size_t)DFF_ * C_);
  conv(fc2_w,  wfc2,  (size_t)C_ * DFF_);

  // 1) LN1
  layernorm_kernel<<<MROWS, 256, 0, stream>>>(x, ln1_w, ln1_b, xn);
  // 2) QKV GEMM (fp32 out for RoPE precision)
  gemm_bf16_kernel<0><<<dim3(NQKV / 128, MROWS / 128), 128, 0, stream>>>(
      xn, wqkv, MROWS, NQKV, C_, qkvF, nullptr, nullptr, nullptr);
  // 3) RoPE + scale + GQA split
  rope_split_kernel<<<(MROWS * NQKV) / 256, 256, 0, stream>>>(qkvF, cosb, sinb, Qb, Kb, Vb);
  // 4) flash attention
  attention_kernel<<<dim3(T_ / 64, B_ * H_), 128, 0, stream>>>(Qb, Kb, Vb, y);
  // 5) proj + residual
  gemm_bf16_kernel<1><<<dim3(C_ / 128, MROWS / 128), 128, 0, stream>>>(
      y, wproj, MROWS, C_, C_, x1, nullptr, nullptr, x);
  // 6) LN2
  layernorm_kernel<<<MROWS, 256, 0, stream>>>(x1, ln2_w, ln2_b, xn2);
  // 7) fc1 + bias + exact GELU
  gemm_bf16_kernel<2><<<dim3(DFF_ / 128, MROWS / 128), 128, 0, stream>>>(
      xn2, wfc1, MROWS, DFF_, C_, nullptr, mlp, fc1_b, nullptr);
  // 8) fc2 + bias + residual -> output
  gemm_bf16_kernel<3><<<dim3(C_ / 128, MROWS / 128), 128, 0, stream>>>(
      mlp, wfc2, MROWS, C_, DFF_, out, nullptr, fc2_b, x1);
}
